// TemporalAttention_58798102282423
// MI455X (gfx1250) — compile-verified, matrix operand order checked by reading, not run on hardware
//
#include <hip/hip_runtime.h>

// ---------------------------------------------------------------------------
// TemporalAttention for MI455X (gfx1250, wave32, WMMA bf16)
//   B=4, T=64, P=256, E=512, H=8, D=64
// Workspace layout (bf16):
//   [0)        Wq,Wk,Wv,Wo converted to bf16   : 4 * 512*512      (2 MB)
//   [..)       q,k,v in [B,P,H,T,D]            : 3 * 33.5M        (192 MB)
//   [..)       o in [B,T,P,E]                  : 33.5M            (64 MB)
// ---------------------------------------------------------------------------

typedef __bf16 bf16;
typedef __attribute__((ext_vector_type(16))) __bf16 v16bf;
typedef __attribute__((ext_vector_type(8)))  __bf16 v8bf;
typedef __attribute__((ext_vector_type(8)))  float  v8f;

#define B_ 4
#define T_ 64
#define P_ 256
#define E_ 512
#define H_ 8
#define D_ 64

// WMMA 16x16x32 bf16 fragment element j holds K = ((j&8)<<1)+(j&7)+8*half.
// Equivalently: two contiguous 8-element runs at K = 8*half and K = 16+8*half.
#define KMAP(j) ((((j) & 8) << 1) + ((j) & 7))

static __device__ inline v8f vzero8() {
    v8f z;
    #pragma unroll
    for (int i = 0; i < 8; ++i) z[i] = 0.0f;
    return z;
}

// Load a 16-element bf16 A/B fragment: two 16-byte vector loads.
// p must point at element K=(k0 + 8*half); second run is +16 elements.
static __device__ inline v16bf load_frag(const bf16* p) {
    const v8bf lo = *(const v8bf*)(p);
    const v8bf hi = *(const v8bf*)(p + 16);
    v16bf r;
    #pragma unroll
    for (int j = 0; j < 8; ++j) { r[j] = lo[j]; r[j + 8] = hi[j]; }
    return r;
}

// ---------------------------------------------------------------------------
// Kernel 0: one-shot f32 -> bf16 conversion of the four weight matrices.
// ---------------------------------------------------------------------------
__global__ __launch_bounds__(256) void cvt_weights_kernel(
    const float* __restrict__ Wq, const float* __restrict__ Wk,
    const float* __restrict__ Wv, const float* __restrict__ Wo,
    bf16* __restrict__ wbf)
{
    const int i = blockIdx.x * 256 + threadIdx.x;   // 0 .. 262143
    const int n = E_ * E_;
    wbf[i]         = (__bf16)Wq[i];
    wbf[n + i]     = (__bf16)Wk[i];
    wbf[2 * n + i] = (__bf16)Wv[i];
    wbf[3 * n + i] = (__bf16)Wo[i];
}

// ---------------------------------------------------------------------------
// Kernel 1: fused QKV projection.
//   x [N=65536, E=512] f32 ;  q/k/v = x @ W^T + b  -> bf16 in [B,P,H,T,D]
// Block: 256 threads (8 waves). Block owns 16 token rows; wave w owns cols
// [w*64, w*64+64). Each wave holds 3 matrices x 4 col-tiles of f32 accum.
// ---------------------------------------------------------------------------
__global__ __launch_bounds__(256) void qkv_proj_kernel(
    const float* __restrict__ x,
    const bf16* __restrict__ Wqb, const bf16* __restrict__ Wkb,
    const bf16* __restrict__ Wvb,
    const float* __restrict__ bq, const float* __restrict__ bk,
    const float* __restrict__ bv,
    bf16* __restrict__ qo, bf16* __restrict__ ko, bf16* __restrict__ vo)
{
    const int lane = threadIdx.x & 31;
    const int wave = threadIdx.x >> 5;
    const int half = lane >> 4;
    const int lcol = lane & 15;
    const int n0   = blockIdx.x * 16;
    const int f0   = wave * 64;

    const bf16*  Ws[3]   = {Wqb, Wkb, Wvb};
    const float* bias[3] = {bq, bk, bv};
    bf16*        outs[3] = {qo, ko, vo};

    v8f acc[3][4];
    #pragma unroll
    for (int m = 0; m < 3; ++m)
        #pragma unroll
        for (int c = 0; c < 4; ++c) acc[m][c] = vzero8();

    for (int kk = 0; kk < 16; ++kk) {
        const int k0 = kk * 32;
        // A fragment: 16 rows of x (f32 -> bf16 on the fly), reused 12x
        v16bf af;
        {
            const float* xp = x + (size_t)(n0 + lcol) * E_ + k0 + 8 * half;
            #pragma unroll
            for (int j = 0; j < 16; ++j) af[j] = (__bf16)xp[KMAP(j)];
        }
        #pragma unroll
        for (int m = 0; m < 3; ++m) {
            #pragma unroll
            for (int c = 0; c < 4; ++c) {
                const int f = f0 + c * 16 + lcol;
                const v16bf bfv =
                    load_frag(Ws[m] + (size_t)f * E_ + k0 + 8 * half);
                acc[m][c] = __builtin_amdgcn_wmma_f32_16x16x32_bf16(
                    false, af, false, bfv, (short)0, acc[m][c], false, false);
            }
        }
    }

    // Write bf16 results into [B,P,H,T,D] layout (token n -> (b,t,p), col f -> (h,d))
    #pragma unroll
    for (int m = 0; m < 3; ++m) {
        #pragma unroll
        for (int c = 0; c < 4; ++c) {
            const int f = f0 + c * 16 + lcol;
            const int h = f >> 6;
            const int d = f & 63;
            const float bb = bias[m][f];
            #pragma unroll
            for (int r = 0; r < 8; ++r) {
                const int n  = n0 + r + 8 * half;
                const int p  = n & (P_ - 1);
                const int bt = n >> 8;          // n / 256
                const int t  = bt & (T_ - 1);
                const int b  = bt >> 6;
                const size_t off =
                    ((((size_t)b * P_ + p) * H_ + h) * T_ + t) * D_ + d;
                outs[m][off] = (__bf16)(acc[m][c][r] + bb);
            }
        }
    }
}

// ---------------------------------------------------------------------------
// Kernel 2: causal attention per (b,p,h). One wave per block.
//   S = QK^T/8 (WMMA, skip fully-masked tiles) -> LDS -> softmax -> bf16 probs
//   O = P @ V (V staged transposed in LDS) -> o [B,T,P,E] bf16
// ---------------------------------------------------------------------------
__global__ __launch_bounds__(32) void attn_kernel(
    const bf16* __restrict__ q, const bf16* __restrict__ k,
    const bf16* __restrict__ v, bf16* __restrict__ o)
{
    __shared__ float  Ssc[T_ * T_];       // 16 KB scores
    __shared__ __bf16 Ap[T_ * T_];        // 8 KB probs (A fragments source)
    __shared__ __bf16 vT[D_ * 72];        // 9 KB V^T, stride 72 keeps 16B align

    const int lane = threadIdx.x;
    const int half = lane >> 4;
    const int lcol = lane & 15;

    const int idx = blockIdx.x;
    const int h = idx & (H_ - 1);
    const int p = (idx >> 3) & (P_ - 1);
    const int b = idx >> 11;

    const size_t base = (((size_t)b * P_ + p) * H_ + h) * (T_ * D_);
    const bf16* qb = q + base;
    const bf16* kb = k + base;
    const bf16* vb = v + base;

    // Stage V^T : vT[d][s] = v[s][d]
    for (int i = lane; i < T_ * D_; i += 32) {
        const int s = i >> 6, d = i & 63;
        vT[d * 72 + s] = vb[i];
    }
    __syncthreads();

    // S = Q K^T * (1/sqrt(D))
    #pragma unroll
    for (int it = 0; it < 4; ++it) {
        #pragma unroll
        for (int is = 0; is < 4; ++is) {
            v8f acc = vzero8();
            if (is <= it) {                 // tiles above diagonal fully masked
                #pragma unroll
                for (int kk = 0; kk < 2; ++kk) {
                    const int d0 = kk * 32 + 8 * half;
                    const v16bf af  = load_frag(qb + (it * 16 + lcol) * D_ + d0);
                    const v16bf bfv = load_frag(kb + (is * 16 + lcol) * D_ + d0);
                    acc = __builtin_amdgcn_wmma_f32_16x16x32_bf16(
                        false, af, false, bfv, (short)0, acc, false, false);
                }
            }
            #pragma unroll
            for (int r = 0; r < 8; ++r)
                Ssc[(it * 16 + r + 8 * half) * T_ + is * 16 + lcol] =
                    acc[r] * 0.125f;
        }
    }
    __syncthreads();

    // Causal softmax: 2 rows per lane
    #pragma unroll
    for (int rr = 0; rr < 2; ++rr) {
        const int t = lane + rr * 32;
        float mx = -3.402823466e38f;
        for (int s = 0; s <= t; ++s) mx = fmaxf(mx, Ssc[t * T_ + s]);
        float sum = 0.0f;
        for (int s = 0; s <= t; ++s) sum += __expf(Ssc[t * T_ + s] - mx);
        const float inv = 1.0f / sum;
        for (int s = 0; s < T_; ++s)
            Ap[t * T_ + s] =
                (__bf16)((s <= t) ? __expf(Ssc[t * T_ + s] - mx) * inv : 0.0f);
    }
    __syncthreads();

    // O = P @ V
    #pragma unroll
    for (int it = 0; it < 4; ++it) {
        #pragma unroll
        for (int dt = 0; dt < 4; ++dt) {
            v8f acc = vzero8();
            #pragma unroll
            for (int kk = 0; kk < 2; ++kk) {
                if (kk * 32 >= (it + 1) * 16) continue;  // probs zero beyond diag
                const int s0 = kk * 32 + 8 * half;
                const v16bf af  = load_frag(&Ap[(it * 16 + lcol) * T_ + s0]);
                const v16bf bfv = load_frag(&vT[(dt * 16 + lcol) * 72 + s0]);
                acc = __builtin_amdgcn_wmma_f32_16x16x32_bf16(
                    false, af, false, bfv, (short)0, acc, false, false);
            }
            #pragma unroll
            for (int r = 0; r < 8; ++r) {
                const int t = it * 16 + r + 8 * half;
                const int d = dt * 16 + lcol;
                const size_t off =
                    (((size_t)b * T_ + t) * P_ + p) * E_ + h * D_ + d;
                o[off] = (__bf16)acc[r];
            }
        }
    }
}

// ---------------------------------------------------------------------------
// Kernel 3: out-projection + residual + LayerNorm, fused.
// Block: 256 threads; computes full 16x512 row panel of y = o@Wo^T + bo + x
// in LDS, then LayerNorm over E=512 and write f32 output.
// ---------------------------------------------------------------------------
__global__ __launch_bounds__(256) void oproj_ln_kernel(
    const bf16* __restrict__ o, const bf16* __restrict__ Wob,
    const float* __restrict__ bo, const float* __restrict__ x,
    const float* __restrict__ lng, const float* __restrict__ lnb,
    float* __restrict__ out)
{
    __shared__ float ysh[16 * E_];        // 32 KB
    __shared__ float redS[16 * 17];
    __shared__ float redQ[16 * 17];
    __shared__ float muv[16];
    __shared__ float rsv[16];

    const int lane = threadIdx.x & 31;
    const int wave = threadIdx.x >> 5;
    const int half = lane >> 4;
    const int lcol = lane & 15;
    const int n0   = blockIdx.x * 16;
    const int f0   = wave * 64;

    v8f acc[4];
    #pragma unroll
    for (int c = 0; c < 4; ++c) acc[c] = vzero8();

    for (int kk = 0; kk < 16; ++kk) {
        const int k0 = kk * 32;
        const v16bf af =
            load_frag(o + (size_t)(n0 + lcol) * E_ + k0 + 8 * half);
        #pragma unroll
        for (int c = 0; c < 4; ++c) {
            const int f = f0 + c * 16 + lcol;
            const v16bf bfv =
                load_frag(Wob + (size_t)f * E_ + k0 + 8 * half);
            acc[c] = __builtin_amdgcn_wmma_f32_16x16x32_bf16(
                false, af, false, bfv, (short)0, acc[c], false, false);
        }
    }

    // y = attn_out + bo + x  -> LDS
    #pragma unroll
    for (int c = 0; c < 4; ++c) {
        const int f = f0 + c * 16 + lcol;
        const float bb = bo[f];
        #pragma unroll
        for (int r = 0; r < 8; ++r) {
            const int m = r + 8 * half;
            const int n = n0 + m;
            ysh[m * E_ + f] = acc[c][r] + bb + x[(size_t)n * E_ + f];
        }
    }
    __syncthreads();

    // LayerNorm over E=512: 16 threads per row, 32 cols each
    const int row  = threadIdx.x >> 4;
    const int tsub = threadIdx.x & 15;
    float s = 0.0f, sq = 0.0f;
    #pragma unroll
    for (int j = 0; j < 32; ++j) {
        const float val = ysh[row * E_ + j * 16 + tsub];
        s += val;
        sq += val * val;
    }
    redS[row * 17 + tsub] = s;
    redQ[row * 17 + tsub] = sq;
    __syncthreads();
    if (tsub == 0) {
        float ts = 0.0f, tq = 0.0f;
        #pragma unroll
        for (int j = 0; j < 16; ++j) {
            ts += redS[row * 17 + j];
            tq += redQ[row * 17 + j];
        }
        const float mu  = ts * (1.0f / E_);
        const float var = tq * (1.0f / E_) - mu * mu;
        muv[row] = mu;
        rsv[row] = rsqrtf(var + 1e-5f);
    }
    __syncthreads();
    const float mu = muv[row], rs = rsv[row];
    #pragma unroll
    for (int j = 0; j < 32; ++j) {
        const int f = j * 16 + tsub;
        const float val = ysh[row * E_ + f];
        out[(size_t)(n0 + row) * E_ + f] = (val - mu) * rs * lng[f] + lnb[f];
    }
}

// ---------------------------------------------------------------------------
extern "C" void kernel_launch(void* const* d_in, const int* in_sizes, int n_in,
                              void* d_out, int out_size, void* d_ws, size_t ws_size,
                              hipStream_t stream) {
    const float* x  = (const float*)d_in[0];
    const float* Wq = (const float*)d_in[1];
    const float* bq = (const float*)d_in[2];
    const float* Wk = (const float*)d_in[3];
    const float* bk = (const float*)d_in[4];
    const float* Wv = (const float*)d_in[5];
    const float* bv = (const float*)d_in[6];
    const float* Wo = (const float*)d_in[7];
    const float* bo = (const float*)d_in[8];
    const float* lg = (const float*)d_in[9];
    const float* lb = (const float*)d_in[10];
    float* out = (float*)d_out;

    const size_t WN = (size_t)E_ * E_;            // 262,144 per weight matrix
    const size_t M  = (size_t)B_ * T_ * P_ * E_;  // 33,554,432 elements

    bf16* wbf = (bf16*)d_ws;                      // 4 weight matrices, bf16
    bf16* Wqb = wbf;
    bf16* Wkb = wbf + WN;
    bf16* Wvb = wbf + 2 * WN;
    bf16* Wob = wbf + 3 * WN;
    bf16* q   = wbf + 4 * WN;                     // [B,P,H,T,D]
    bf16* k   = q + M;
    bf16* v   = k + M;
    bf16* o   = v + M;                            // [B,T,P,E]

    const int N = B_ * T_ * P_;                   // 65536 tokens

    cvt_weights_kernel<<<(int)(WN / 256), 256, 0, stream>>>(Wq, Wk, Wv, Wo, wbf);
    qkv_proj_kernel<<<N / 16, 256, 0, stream>>>(x, Wqb, Wkb, Wvb, bq, bk, bv,
                                                q, k, v);
    attn_kernel<<<B_ * P_ * H_, 32, 0, stream>>>(q, k, v, o);
    oproj_ln_kernel<<<N / 16, 256, 0, stream>>>(o, Wob, bo, x, lg, lb, out);
}
